// RowTriangleAttentionBlock_44727789420923
// MI455X (gfx1250) — matrix operand order repaired and checked by reading, not compiled, not run on hardware
//
#include <hip/hip_runtime.h>

typedef __attribute__((ext_vector_type(16))) __bf16 v16bf;
typedef __attribute__((ext_vector_type(8)))  float  v8f;
typedef unsigned int u32;

#define NROW 320
#define CDIM 128
#define DDIM 64
#define HHEAD 4
#define CH 32
#define MROWS (NROW * NROW)   // 102400 flattened (i,j) rows

// ---------- wave reductions (wave32) ----------
__device__ __forceinline__ float wave_max32(float v) {
    #pragma unroll
    for (int o = 16; o > 0; o >>= 1) v = fmaxf(v, __shfl_xor(v, o, 32));
    return v;
}
__device__ __forceinline__ float wave_sum32(float v) {
    #pragma unroll
    for (int o = 16; o > 0; o >>= 1) v += __shfl_xor(v, o, 32);
    return v;
}

// LDS byte offset of a generic pointer (for async-to-LDS VDST operand)
__device__ __forceinline__ unsigned lds_addr_of(const void* p) {
    return (unsigned)(unsigned long long)(const __attribute__((address_space(3))) char*)p;
}

// ---------- WMMA fragment loaders (bf16, 16x16x32) ----------
// A matrix 16x32 (MxK), this code's K order (not the ISA 7.12.2 table order):
//  vgpr 0..7 : K = 0..15 (lanes 0-15) / 16..31 (lanes 16-31), K contiguous (matches load_B_kcontig)
__device__ __forceinline__ v16bf load_A_frag(const __bf16* base, int stride, int k0, int lane) {
    int m = lane & 15, hi = lane >> 4;
    const __bf16* rp = base + (long)m * stride + k0 + 16 * hi;
    union { v16bf v; u32 w[8]; } a;
    #pragma unroll
    for (int ep = 0; ep < 8; ++ep) {
        int col = 2 * ep;
        a.w[ep] = *(const u32*)(rp + col);
    }
    return a.v;
}

// B matrix 32x16 (KxN): lanes 0-15 -> N, K=0..15 across vgprs 0..7; lanes 16-31: K=16..31.
// K contiguous in memory per lane: element (k,n) at base + (n0+n)*nstride + k.
__device__ __forceinline__ v16bf load_B_kcontig(const __bf16* base, int nstride, int n0, int k0, int lane) {
    int n = lane & 15, hi = lane >> 4;
    const __bf16* p = base + (long)(n0 + n) * nstride + k0 + 16 * hi;
    union { v16bf v; u32 w[8]; } b;
    #pragma unroll
    for (int ep = 0; ep < 8; ++ep) b.w[ep] = *(const u32*)(p + 2 * ep);
    return b.v;
}

// ---------- kernel 1: f32 weights -> bf16 transposed [n][k] ----------
__global__ void k_wconv(const float* Wq, const float* Wk, const float* Wv,
                        const float* Wg, const float* Wo, __bf16* wt) {
    int t = blockIdx.x * blockDim.x + threadIdx.x;
    if (t >= 5 * CDIM * CDIM) return;
    int wi = t >> 14, idx = t & (CDIM * CDIM - 1);
    int nrow = idx >> 7, kcol = idx & 127;
    const float* W = (wi == 0) ? Wq : (wi == 1) ? Wk : (wi == 2) ? Wv : (wi == 3) ? Wg : Wo;
    wt[t] = (__bf16)W[kcol * CDIM + nrow];
}

// ---------- kernel 2: dist bias  Dbias[h][q][k] = (de[k,q]·Wd+bd)h * sig((de[k,q]·Wdg+bdg)h)
__global__ void k_dbias(const float* de, const float* Wd, const float* bd,
                        const float* Wdg, const float* bdg, float* dbias) {
    int t = blockIdx.x * blockDim.x + threadIdx.x;
    if (t >= MROWS) return;
    int kcol = t / NROW, q = t - kcol * NROW;
    const float* e = de + ((long)kcol * NROW + q) * DDIM;
    float a[HHEAD] = {0, 0, 0, 0}, g[HHEAD] = {0, 0, 0, 0};
    #pragma unroll 4
    for (int d = 0; d < DDIM; ++d) {
        float x = e[d];
        #pragma unroll
        for (int h = 0; h < HHEAD; ++h) {
            a[h] += x * Wd[d * HHEAD + h];
            g[h] += x * Wdg[d * HHEAD + h];
        }
    }
    #pragma unroll
    for (int h = 0; h < HHEAD; ++h) {
        float s = 1.0f / (1.0f + __expf(-(g[h] + bdg[h])));
        dbias[((long)h * NROW + q) * NROW + kcol] = (a[h] + bd[h]) * s;
    }
}

// ---------- kernel 3: LayerNorm + q/k/v/g projections (WMMA) ----------
// 128 threads (4 waves), 64 rows of the flattened [102400,128] activation per block.
__global__ void k_ln_proj(const float* xin, const float* ln_w, const float* ln_b,
                          const __bf16* wt, const float* bg,
                          __bf16* qb, __bf16* kb, __bf16* vb, __bf16* gb) {
    __shared__ __bf16 xs[64][CDIM];
    int lane = threadIdx.x & 31, w = threadIdx.x >> 5;
    long g0 = (long)blockIdx.x * 64;

    if (blockIdx.x + 1 < gridDim.x)
        __builtin_prefetch(xin + (g0 + 64) * CDIM + threadIdx.x, 0, 0);

    // LayerNorm: wave w owns rows w*16 .. w*16+15 (32 lanes x 4 cols each)
    float w0 = ln_w[lane], w1 = ln_w[lane + 32], w2 = ln_w[lane + 64], w3 = ln_w[lane + 96];
    float b0 = ln_b[lane], b1 = ln_b[lane + 32], b2 = ln_b[lane + 64], b3 = ln_b[lane + 96];
    for (int r = 0; r < 16; ++r) {
        int row = w * 16 + r;
        const float* xr = xin + (g0 + row) * CDIM;
        float v0 = xr[lane], v1 = xr[lane + 32], v2 = xr[lane + 64], v3 = xr[lane + 96];
        float mu = wave_sum32(v0 + v1 + v2 + v3) * (1.0f / 128.0f);
        float d0 = v0 - mu, d1 = v1 - mu, d2 = v2 - mu, d3 = v3 - mu;
        float var = wave_sum32(d0 * d0 + d1 * d1 + d2 * d2 + d3 * d3) * (1.0f / 128.0f);
        float inv = rsqrtf(var + 1e-5f);
        xs[row][lane]      = (__bf16)(d0 * inv * w0 + b0);
        xs[row][lane + 32] = (__bf16)(d1 * inv * w1 + b1);
        xs[row][lane + 64] = (__bf16)(d2 * inv * w2 + b2);
        xs[row][lane + 96] = (__bf16)(d3 * inv * w3 + b3);
    }
    // each wave consumes only its own 16 rows -> no barrier needed

    // hoist A fragments (16x128 per wave = 4 K-steps); xs becomes reusable after this
    v16bf afr[4];
    #pragma unroll
    for (int kt = 0; kt < 4; ++kt) afr[kt] = load_A_frag(&xs[w * 16][0], CDIM, kt * 32, lane);

    __bf16* outs[4] = {qb, kb, vb, gb};
    int nlo = lane & 15, hi = lane >> 4;
    for (int wi = 0; wi < 4; ++wi) {
        const __bf16* Wt = wt + (long)wi * CDIM * CDIM; // transposed [n][k]
        #pragma unroll
        for (int nt = 0; nt < 8; ++nt) {
            v8f acc = {};
            #pragma unroll
            for (int kt = 0; kt < 4; ++kt) {
                v16bf bfrag = load_B_kcontig(Wt, CDIM, nt * 16, kt * 32, lane);
                acc = __builtin_amdgcn_wmma_f32_16x16x32_bf16(
                        false, afr[kt], false, bfrag, (short)0, acc, false, false);
            }
            int col = nt * 16 + nlo;
            #pragma unroll
            for (int r = 0; r < 8; ++r) {
                int m = r + 8 * hi;
                float val = acc[r];
                if (wi == 0) val *= 0.17677669529663687f;                    // 1/sqrt(CH)
                if (wi == 3) val = 1.0f / (1.0f + __expf(-(val + bg[col]))); // gate
                xs[w * 16 + m][col] = (__bf16)val;   // repack into wave-private LDS region
            }
        }
        // coalesced b128 stores: 16 rows x 256B contiguous in global
        __bf16* dstg = outs[wi] + (g0 + w * 16) * CDIM;
        const uint4* lsrc = (const uint4*)&xs[w * 16][0];
        #pragma unroll
        for (int c = lane; c < 256; c += 32) ((uint4*)dstg)[c] = lsrc[c];
    }
}

// ---------- kernel 4: attention per (i, h, q-tile of 32) ----------
// 64 threads (2 waves). QK^T (one WMMA K-step, CH=32) -> bias -> softmax -> P@V -> gate.
__global__ void k_attn(const __bf16* qb, const __bf16* kb, const __bf16* vb, const __bf16* gb,
                       const float* pmask, const float* dbias, __bf16* og) {
    __shared__ float  lg[32][NROW];    // 40 KB logits (later reused as V^T bf16)
    __shared__ __bf16 pr[32][NROW];    // 20 KB probs (bf16)
    __shared__ __bf16 ot[32][CH];      //  2 KB gated output repack tile
    int lane = threadIdx.x & 31, w = threadIdx.x >> 5;
    int qt = blockIdx.x, h = blockIdx.y, i = blockIdx.z;
    int q0 = qt * 32;
    long base = (long)i * NROW * CDIM + h * CH;

    // Q A-fragment: rows q0+w*16 .. +15, cols h*CH..+31 (K=32 == CH)
    v16bf aq = load_A_frag(qb + base + (long)(q0 + w * 16) * CDIM, CDIM, 0, lane);

    int nlo = lane & 15, hi = lane >> 4;
    const float* mrow = pmask + (long)i * NROW;
    const float* drow = dbias + ((long)h * NROW + (q0 + w * 16)) * NROW;

    for (int nt = 0; nt < 20; ++nt) {
        // B = K^T: element (kk, n) = kb[(i*320 + j0+n)*128 + h*32 + kk]  (K contiguous)
        v16bf bk = load_B_kcontig(kb + base, CDIM, nt * 16, 0, lane);
        v8f acc = {};
        acc = __builtin_amdgcn_wmma_f32_16x16x32_bf16(false, aq, false, bk, (short)0, acc, false, false);
        int j = nt * 16 + nlo;
        float mb = 1000000000.0f * (mrow[j] - 1.0f);
        #pragma unroll
        for (int r = 0; r < 8; ++r) {
            int m = r + 8 * hi;
            lg[w * 16 + m][j] = acc[r] + mb + drow[(long)m * NROW + j];
        }
    }

    // softmax over k (each wave: its own 16 rows, 32 lanes strided over 320 cols)
    for (int r = 0; r < 16; ++r) {
        int row = w * 16 + r;
        float mx = -3.4e38f;
        for (int c = lane; c < NROW; c += 32) mx = fmaxf(mx, lg[row][c]);
        mx = wave_max32(mx);
        float s = 0.0f;
        for (int c = lane; c < NROW; c += 32) {
            float e = __expf(lg[row][c] - mx);
            lg[row][c] = e;
            s += e;
        }
        s = wave_sum32(s);
        float inv = 1.0f / s;
        for (int c = lane; c < NROW; c += 32) pr[row][c] = (__bf16)(lg[row][c] * inv);
    }
    __syncthreads();                       // lg (f32 logits) now dead everywhere

    // stage V^T into LDS reusing lg: vt[ch][j] = v[i, j, h*32+ch]  (20 KB of the 40)
    __bf16* vt = (__bf16*)&lg[0][0];
    const __bf16* vsrc = vb + base;
    for (int c = threadIdx.x; c < NROW * 4; c += blockDim.x) {
        int j = c >> 2, ch0 = (c & 3) * 8;
        uint4 d = *(const uint4*)(vsrc + (long)j * CDIM + ch0);
        const __bf16* dv = (const __bf16*)&d;
        #pragma unroll
        for (int e = 0; e < 8; ++e) vt[(ch0 + e) * NROW + j] = dv[e];
    }
    __syncthreads();

    // O = P @ V  (K = 320 -> 10 WMMA steps, N = 32 -> 2 tiles)
    v8f accs[2];
    #pragma unroll
    for (int nt2 = 0; nt2 < 2; ++nt2) {
        v8f acc = {};
        for (int kt = 0; kt < 10; ++kt) {
            v16bf a = load_A_frag(&pr[w * 16][0], NROW, kt * 32, lane);
            v16bf b = load_B_kcontig(vt, NROW, nt2 * 16, kt * 32, lane);
            acc = __builtin_amdgcn_wmma_f32_16x16x32_bf16(false, a, false, b, (short)0, acc, false, false);
        }
        accs[nt2] = acc;
    }
    // gate with g and repack into wave-private LDS tile
    #pragma unroll
    for (int nt2 = 0; nt2 < 2; ++nt2) {
        #pragma unroll
        for (int r = 0; r < 8; ++r) {
            int m = r + 8 * hi;
            long off = base + (long)(q0 + w * 16 + m) * CDIM + nt2 * 16 + nlo;
            ot[w * 16 + m][nt2 * 16 + nlo] = (__bf16)(accs[nt2][r] * (float)gb[off]);
        }
    }
    // coalesced b128 stores of the 32x32 gated tile (2 chunks per lane)
    #pragma unroll
    for (int cc = 0; cc < 2; ++cc) {
        int c = cc * 32 + lane;            // 64 chunks of 16B over the wave's 16 rows
        int row = c >> 2, seg = c & 3;     // row 0..15 within wave region, seg*8 bf16
        __bf16* gdst = og + base + (long)(q0 + w * 16 + row) * CDIM + seg * 8;
        *(uint4*)gdst = *(const uint4*)&ot[w * 16 + row][seg * 8];
    }
}

// ---------- kernel 5: output projection + bias + mask + residual ----------
__global__ void k_outproj(const __bf16* og, const __bf16* wt, const float* bo,
                          const float* xin, const float* pmask, float* out) {
    __shared__ __bf16 os[64][CDIM];
    int lane = threadIdx.x & 31, w = threadIdx.x >> 5;
    long g0 = (long)blockIdx.x * 64;
    const __bf16* Wto = wt + (long)4 * CDIM * CDIM;

    if (blockIdx.x + 1 < gridDim.x)
        __builtin_prefetch(og + (g0 + 64) * CDIM + threadIdx.x, 0, 0);

    // stage 64x128 bf16 tile via gfx1250 async copy (ASYNCcnt path):
    // 1024 x 16B chunks, 128 threads -> 8 async b128 issues per thread
    {
        unsigned lbase = lds_addr_of(&os[0][0]);
        const char* gsrc = (const char*)(og + g0 * CDIM);
        for (int c = threadIdx.x; c < 64 * CDIM * 2 / 16; c += blockDim.x) {
            unsigned laddr = lbase + c * 16;
            const char* ga = gsrc + (long)c * 16;
            asm volatile("global_load_async_to_lds_b128 %0, %1, off"
                         :: "v"(laddr), "v"(ga) : "memory");
        }
        asm volatile("s_wait_asynccnt 0x0" ::: "memory");
    }
    __syncthreads();

    v16bf afr[4];
    #pragma unroll
    for (int kt = 0; kt < 4; ++kt) afr[kt] = load_A_frag(&os[w * 16][0], CDIM, kt * 32, lane);

    int nlo = lane & 15, hi = lane >> 4;
    #pragma unroll
    for (int nt = 0; nt < 8; ++nt) {
        v8f acc = {};
        #pragma unroll
        for (int kt = 0; kt < 4; ++kt) {
            v16bf bfrag = load_B_kcontig(Wto, CDIM, nt * 16, kt * 32, lane);
            acc = __builtin_amdgcn_wmma_f32_16x16x32_bf16(
                    false, afr[kt], false, bfrag, (short)0, acc, false, false);
        }
        int col = nt * 16 + nlo;
        float bias = bo[col];
        #pragma unroll
        for (int r = 0; r < 8; ++r) {
            int m = r + 8 * hi;
            long row = g0 + w * 16 + m;
            out[row * CDIM + col] = xin[row * CDIM + col] + (acc[r] + bias) * pmask[row];
        }
    }
}

extern "C" void kernel_launch(void* const* d_in, const int* in_sizes, int n_in,
                              void* d_out, int out_size, void* d_ws, size_t ws_size,
                              hipStream_t stream) {
    const float* pair_embed = (const float*)d_in[0];
    const float* pair_mask  = (const float*)d_in[1];
    const float* dist_embed = (const float*)d_in[2];
    const float* ln_w = (const float*)d_in[3];
    const float* ln_b = (const float*)d_in[4];
    const float* Wd   = (const float*)d_in[5];
    const float* bd   = (const float*)d_in[6];
    const float* Wdg  = (const float*)d_in[7];
    const float* bdg  = (const float*)d_in[8];
    const float* Wq   = (const float*)d_in[9];
    const float* Wk   = (const float*)d_in[10];
    const float* Wv   = (const float*)d_in[11];
    const float* Wg   = (const float*)d_in[12];
    const float* bg   = (const float*)d_in[13];
    const float* Wo   = (const float*)d_in[14];
    const float* bo   = (const float*)d_in[15];
    float* out = (float*)d_out;

    // workspace carve (bf16 activations halve intermediate HBM traffic)
    char* p = (char*)d_ws;
    __bf16* wt = (__bf16*)p;           p += (size_t)5 * CDIM * CDIM * 2;       // 160 KB
    __bf16* qb = (__bf16*)p;           p += (size_t)MROWS * CDIM * 2;          // 26.2 MB
    __bf16* kb = (__bf16*)p;           p += (size_t)MROWS * CDIM * 2;
    __bf16* vb = (__bf16*)p;           p += (size_t)MROWS * CDIM * 2;
    __bf16* gb = (__bf16*)p;           p += (size_t)MROWS * CDIM * 2;
    __bf16* og = (__bf16*)p;           p += (size_t)MROWS * CDIM * 2;
    float*  dbias = (float*)p;         p += (size_t)HHEAD * NROW * NROW * 4;   // 1.6 MB

    k_wconv<<<(5 * CDIM * CDIM + 255) / 256, 256, 0, stream>>>(Wq, Wk, Wv, Wg, Wo, wt);
    k_dbias<<<(MROWS + 255) / 256, 256, 0, stream>>>(dist_embed, Wd, bd, Wdg, bdg, dbias);
    k_ln_proj<<<MROWS / 64, 128, 0, stream>>>(pair_embed, ln_w, ln_b, wt, bg, qb, kb, vb, gb);
    k_attn<<<dim3(NROW / 32, HHEAD, NROW), 64, 0, stream>>>(qb, kb, vb, gb, pair_mask, dbias, og);
    k_outproj<<<MROWS / 64, 128, 0, stream>>>(og, wt, bo, pair_embed, pair_mask, out);
}
